// GAT_82085414961435
// MI455X (gfx1250) — compile-verified, matrix operand order checked by reading, not run on hardware
//
#include <hip/hip_runtime.h>
#include <hip/hip_bf16.h>
#include <hip/hip_fp16.h>

#define NNODE 50000
#define NEDGE 800000
#define EMB   64
#define NH    4
#define CH    16
#define NLAYER 3
#define NEGS  0.2f
#define BNEPS 1e-5f

typedef __attribute__((ext_vector_type(16))) _Float16 v16h;
typedef __attribute__((ext_vector_type(8)))  _Float16 v8h;
typedef __attribute__((ext_vector_type(8)))  float    v8f;

// ---- order-preserving float <-> uint encoding for atomicMax on floats ----
__device__ __forceinline__ unsigned f2ord(float f) {
  unsigned b = __float_as_uint(f);
  return (b & 0x80000000u) ? ~b : (b | 0x80000000u);
}
__device__ __forceinline__ float ord2f(unsigned k) {
  unsigned b = (k & 0x80000000u) ? (k ^ 0x80000000u) : ~k;
  return __uint_as_float(b);
}

// ---- x = atom_emb[x_idx]; also emit f16 copy for the WMMA GEMM ----
__global__ void embed_kernel(const int* __restrict__ xidx,
                             const float* __restrict__ atom_emb,
                             float* __restrict__ x, _Float16* __restrict__ xh) {
  int tid = blockIdx.x * blockDim.x + threadIdx.x;
  if (tid >= NNODE * EMB) return;
  int n = tid >> 6, c = tid & 63;
  float v = atom_emb[(size_t)xidx[n] * EMB + c];
  x[tid] = v;
  xh[tid] = (_Float16)v;
}

// ---- pack W[64x64] (f32, k-major) into WMMA B-operand lane layout (f16) ----
// Bp index: (((ks*4 + nt)*32 + lane)*16 + j); lane -> N = nt*16 + lane%16,
// g = lane/16; j<8 -> k = ks*32 + g*8 + j; j>=8 -> k = ks*32 + 16 + g*8 + (j-8).
__global__ void packW_kernel(const float* __restrict__ W, _Float16* __restrict__ Bp) {
  int tid = blockIdx.x * blockDim.x + threadIdx.x;
  if (tid >= 4096) return;
  int j = tid & 15, lane = (tid >> 4) & 31, nt = (tid >> 9) & 3, ks = tid >> 11;
  int g = lane >> 4;
  int n = nt * 16 + (lane & 15);
  int k = ks * 32 + ((j < 8) ? (g * 8 + j) : (16 + g * 8 + (j - 8)));
  Bp[tid] = (_Float16)W[k * EMB + n];
}

// ---- xl = x @ W via v_wmma_f32_16x16x32_f16 --------------------------------
// One wave per 16-row tile; 4 column tiles; K=64 in two 32-wide steps.
// B comes pre-packed: one contiguous 32B load per (ks,nt) tile per lane.
__global__ void gemm_xw_wmma(const _Float16* __restrict__ A,   // [nrows,64]
                             const _Float16* __restrict__ Bp,  // packed [2][4][32][16]
                             float* __restrict__ D, int nrows) {
  int wv = (int)((blockIdx.x * (unsigned)blockDim.x + threadIdx.x) >> 5);
  int nwaves = nrows >> 4;
  if (wv >= nwaves) return;                   // wave-uniform: EXEC stays all-ones
  int lane = threadIdx.x & 31;
  int g = lane >> 4, mr = lane & 15;
  v8f acc[4] = {};
  #pragma unroll
  for (int ks = 0; ks < 2; ++ks) {
    const _Float16* ar = A + (size_t)(wv * 16 + mr) * EMB + ks * 32 + g * 8;
    v8h alo = *(const v8h*)ar;            // 16B aligned
    v8h ahi = *(const v8h*)(ar + 16);     // 16B aligned
    v16h a = __builtin_shufflevector(alo, ahi, 0, 1, 2, 3, 4, 5, 6, 7,
                                     8, 9, 10, 11, 12, 13, 14, 15);
    #pragma unroll
    for (int nt = 0; nt < 4; ++nt) {
      v16h b = *(const v16h*)(Bp + (((ks * 4 + nt) * 32 + lane) << 4));  // 32B aligned
      acc[nt] = __builtin_amdgcn_wmma_f32_16x16x32_f16(
          false, a, false, b, (short)0, acc[nt], false, false);
    }
  }
  #pragma unroll
  for (int nt = 0; nt < 4; ++nt)
    #pragma unroll
    for (int r = 0; r < 8; ++r)
      D[(size_t)(wv * 16 + g * 8 + r) * EMB + nt * 16 + mr] = acc[nt][r];
}

// ---- aet[t][h] = sum_c edge_emb[t][h*16+c] * att_edge[h][c] (4x4 table) ----
__global__ void aet_kernel(const float* __restrict__ eemb,   // [4,64]
                           const float* __restrict__ attE,   // [H*C]
                           float* __restrict__ aet) {        // [4,H]
  int id = threadIdx.x;
  if (id >= 16) return;
  int t = id >> 2, h = id & 3;
  float s = 0.f;
  for (int c = 0; c < CH; ++c) s += eemb[t * EMB + h * CH + c] * attE[h * CH + c];
  aet[t * NH + h] = s;
}

// ---- per-node histogram of incoming edge-attr types ----
__global__ void hist_kernel(const int* __restrict__ dst, const int* __restrict__ eattr,
                            int* __restrict__ hist) {
  int e = blockIdx.x * blockDim.x + threadIdx.x;
  if (e >= NEDGE) return;
  atomicAdd(&hist[(size_t)dst[e] * 4 + eattr[e]], 1);
}

// ---- per-node: a_src, a_dst, a_loop (self-loop edge score from histogram) ----
__global__ void node_attn(const float* __restrict__ xl,
                          const float* __restrict__ attS, const float* __restrict__ attD,
                          const int* __restrict__ hist, const float* __restrict__ aet,
                          float* __restrict__ a_src, float* __restrict__ a_dst,
                          float* __restrict__ a_loop) {
  int id = blockIdx.x * blockDim.x + threadIdx.x;
  if (id >= NNODE * NH) return;
  int n = id >> 2, h = id & 3;
  const float* xr = xl + (size_t)n * EMB + h * CH;
  float ss = 0.f, sd = 0.f;
  for (int c = 0; c < CH; ++c) {
    ss += xr[c] * attS[h * CH + c];
    sd += xr[c] * attD[h * CH + c];
  }
  a_src[id] = ss;
  a_dst[id] = sd;
  float cnt = 0.f, al = 0.f;
  for (int t = 0; t < 4; ++t) {
    int ht = hist[(size_t)n * 4 + t];
    cnt += (float)ht;
    al += (float)ht * aet[t * NH + h];
  }
  a_loop[id] = al / fmaxf(cnt, 1.f);
}

// ---- alpha (leaky-relu) + running segment max over dst ----
__global__ void alpha_max(const int* __restrict__ src, const int* __restrict__ dst,
                          const int* __restrict__ eattr,
                          const float* __restrict__ a_src, const float* __restrict__ a_dst,
                          const float* __restrict__ a_loop, const float* __restrict__ aet,
                          float* __restrict__ alpha_buf, unsigned* __restrict__ m_enc) {
  int i = blockIdx.x * blockDim.x + threadIdx.x;
  if (i >= NEDGE + NNODE) return;
  int s, d;
  float aev[NH];
  if (i < NEDGE) {
    s = src[i]; d = dst[i];
    int t = eattr[i];
    #pragma unroll
    for (int h = 0; h < NH; ++h) aev[h] = aet[t * NH + h];
  } else {
    s = d = i - NEDGE;
    #pragma unroll
    for (int h = 0; h < NH; ++h) aev[h] = a_loop[(size_t)s * NH + h];
  }
  #pragma unroll
  for (int h = 0; h < NH; ++h) {
    float a = a_src[(size_t)s * NH + h] + a_dst[(size_t)d * NH + h] + aev[h];
    a = (a >= 0.f) ? a : NEGS * a;
    alpha_buf[(size_t)i * NH + h] = a;
    atomicMax(&m_enc[(size_t)d * NH + h], f2ord(a));
  }
}

// ---- ex = exp(alpha - m[dst]); segment-sum denom ----
__global__ void exp_sum(const int* __restrict__ dst, float* __restrict__ alpha_buf,
                        const unsigned* __restrict__ m_enc, float* __restrict__ denom) {
  int i = blockIdx.x * blockDim.x + threadIdx.x;
  if (i >= NEDGE + NNODE) return;
  int d = (i < NEDGE) ? dst[i] : (i - NEDGE);
  #pragma unroll
  for (int h = 0; h < NH; ++h) {
    float m = ord2f(m_enc[(size_t)d * NH + h]);
    float ex = __expf(alpha_buf[(size_t)i * NH + h] - m);
    alpha_buf[(size_t)i * NH + h] = ex;
    atomicAdd(&denom[(size_t)d * NH + h], ex);
  }
}

// ---- msg = alpha * (xl[src] + ea); scatter-add into agg[dst] ----
// 16 threads per (edge|loop) row, 4 contiguous channels each (float4 loads).
__global__ void message(const int* __restrict__ src, const int* __restrict__ dst,
                        const int* __restrict__ eattr, const float* __restrict__ alpha_buf,
                        const float* __restrict__ denom, const float* __restrict__ xl,
                        const float* __restrict__ eemb, const int* __restrict__ hist,
                        float* __restrict__ agg) {
  size_t tid = (size_t)blockIdx.x * blockDim.x + threadIdx.x;
  size_t total = (size_t)(NEDGE + NNODE) * 16;
  if (tid >= total) return;
  int i = (int)(tid >> 4);
  int q = (int)(tid & 15);
  int c0 = q * 4;
  int h = q >> 2;
  int s, d;
  float4 ea4;
  if (i < NEDGE) {
    s = src[i]; d = dst[i];
    ea4 = *(const float4*)(eemb + (size_t)eattr[i] * EMB + c0);
  } else {
    s = d = i - NEDGE;
    float cnt = 0.f;
    float4 acc = make_float4(0.f, 0.f, 0.f, 0.f);
    #pragma unroll
    for (int t = 0; t < 4; ++t) {
      float ht = (float)hist[(size_t)s * 4 + t];
      cnt += ht;
      float4 e = *(const float4*)(eemb + t * EMB + c0);
      acc.x += ht * e.x; acc.y += ht * e.y; acc.z += ht * e.z; acc.w += ht * e.w;
    }
    float inv = 1.f / fmaxf(cnt, 1.f);
    ea4 = make_float4(acc.x * inv, acc.y * inv, acc.z * inv, acc.w * inv);
  }
  float w = alpha_buf[(size_t)i * NH + h] / denom[(size_t)d * NH + h];
  float4 xv = *(const float4*)(xl + (size_t)s * EMB + c0);
  float* ag = agg + (size_t)d * EMB + c0;
  atomicAdd(ag + 0, w * (xv.x + ea4.x));
  atomicAdd(ag + 1, w * (xv.y + ea4.y));
  atomicAdd(ag + 2, w * (xv.z + ea4.z));
  atomicAdd(ag + 3, w * (xv.w + ea4.w));
}

// ---- batchnorm stats: per-channel sum / sumsq with LDS tree reduction ----
__global__ void bn_stats(const float* __restrict__ agg, const float* __restrict__ bias,
                         float* __restrict__ bnsum, float* __restrict__ bnsumsq) {
  int c = blockIdx.x;
  int row = blockIdx.y * 256 + threadIdx.x;
  float v = 0.f;
  if (row < NNODE) v = agg[(size_t)row * EMB + c] + bias[c];
  __shared__ float s1[256];
  __shared__ float s2[256];
  s1[threadIdx.x] = v;
  s2[threadIdx.x] = v * v;
  __syncthreads();
  for (int o = 128; o > 0; o >>= 1) {
    if (threadIdx.x < o) {
      s1[threadIdx.x] += s1[threadIdx.x + o];
      s2[threadIdx.x] += s2[threadIdx.x + o];
    }
    __syncthreads();
  }
  if (threadIdx.x == 0) {
    atomicAdd(&bnsum[c], s1[0]);
    atomicAdd(&bnsumsq[c], s2[0]);
  }
}

// ---- normalize + relu + residual; update x and its f16 shadow ----
__global__ void bn_apply(const float* __restrict__ agg, const float* __restrict__ bias,
                         const float* __restrict__ bnsum, const float* __restrict__ bnsumsq,
                         const float* __restrict__ gamma, const float* __restrict__ beta,
                         float* __restrict__ x, _Float16* __restrict__ xh) {
  int tid = blockIdx.x * blockDim.x + threadIdx.x;
  if (tid >= NNODE * EMB) return;
  int c = tid & 63;
  const float invN = 1.f / (float)NNODE;
  float mean = bnsum[c] * invN;
  float var = bnsumsq[c] * invN - mean * mean;
  float o = (agg[tid] + bias[c] - mean) * rsqrtf(var + BNEPS) * gamma[c] + beta[c];
  float nx = fmaxf(o, 0.f) + x[tid];
  x[tid] = nx;
  xh[tid] = (_Float16)nx;
}

// ---- final edge predictor: [x[src]||x[dst]] @ pred_W + b (float4 dot) ----
__global__ void predict(const float* __restrict__ x, const int* __restrict__ src,
                        const int* __restrict__ dst, const float* __restrict__ pW,
                        const float* __restrict__ pb, float* __restrict__ out) {
  int e = blockIdx.x * blockDim.x + threadIdx.x;
  if (e >= NEDGE) return;
  const float4* xs = (const float4*)(x + (size_t)src[e] * EMB);
  const float4* xd = (const float4*)(x + (size_t)dst[e] * EMB);
  const float4* w1 = (const float4*)pW;
  const float4* w2 = (const float4*)(pW + EMB);
  float acc = pb[0];
  #pragma unroll
  for (int k = 0; k < EMB / 4; ++k) {
    float4 a = xs[k], b = w1[k];
    acc += a.x * b.x + a.y * b.y + a.z * b.z + a.w * b.w;
    float4 cc = xd[k], dd = w2[k];
    acc += cc.x * dd.x + cc.y * dd.y + cc.z * dd.z + cc.w * dd.w;
  }
  out[e] = acc;
}

extern "C" void kernel_launch(void* const* d_in, const int* in_sizes, int n_in,
                              void* d_out, int out_size, void* d_ws, size_t ws_size,
                              hipStream_t stream) {
  const int*   x_idx    = (const int*)d_in[0];
  const int*   eidx     = (const int*)d_in[1];
  const int*   src      = eidx;
  const int*   dst      = eidx + NEDGE;
  const int*   eattr    = (const int*)d_in[2];
  const float* atom_emb = (const float*)d_in[3];
  const float* W        = (const float*)d_in[4];
  const float* attS     = (const float*)d_in[5];
  const float* attD     = (const float*)d_in[6];
  const float* attE     = (const float*)d_in[7];
  const float* bias     = (const float*)d_in[8];
  const float* eembL    = (const float*)d_in[9];
  const float* gamma    = (const float*)d_in[10];
  const float* beta     = (const float*)d_in[11];
  const float* pW       = (const float*)d_in[12];
  const float* pb       = (const float*)d_in[13];
  float* out = (float*)d_out;

  char* ws = (char*)d_ws;
  size_t off = 0;
  auto alloc = [&](size_t bytes) -> char* {
    char* p = ws + off;
    off = (off + bytes + 255) & ~(size_t)255;
    return p;
  };
  float*    x         = (float*)alloc((size_t)NNODE * EMB * 4);
  _Float16* xh        = (_Float16*)alloc((size_t)NNODE * EMB * 2);
  _Float16* Bp        = (_Float16*)alloc((size_t)EMB * EMB * 2);
  float*    xl        = (float*)alloc((size_t)NNODE * EMB * 4);
  float*    agg       = (float*)alloc((size_t)NNODE * EMB * 4);
  float*    alpha_buf = (float*)alloc((size_t)(NEDGE + NNODE) * NH * 4);
  float*    a_src     = (float*)alloc((size_t)NNODE * NH * 4);
  float*    a_dst     = (float*)alloc((size_t)NNODE * NH * 4);
  float*    a_loop    = (float*)alloc((size_t)NNODE * NH * 4);
  int*      hist      = (int*)alloc((size_t)NNODE * 4 * 4);
  unsigned* m_enc     = (unsigned*)alloc((size_t)NNODE * NH * 4);
  float*    denom     = (float*)alloc((size_t)NNODE * NH * 4);
  float*    aet       = (float*)alloc(4 * NH * 4);
  float*    bnsum     = (float*)alloc(EMB * 4);
  float*    bnsumsq   = (float*)alloc(EMB * 4);

  const int TB = 256;
  int gNE = (NNODE * EMB + TB - 1) / TB;          // 12500
  int gEL = (NEDGE + NNODE + TB - 1) / TB;        // 3321
  int gE  = (NEDGE + TB - 1) / TB;                // 3125
  int gNH = (NNODE * NH + TB - 1) / TB;           // 782
  size_t msgTotal = (size_t)(NEDGE + NNODE) * 16;
  int gMsg = (int)((msgTotal + TB - 1) / TB);     // 53125
  int waves = NNODE / 16;                          // 3125 (exact)
  int gW = (waves * 32 + 127) / 128;               // 782 blocks of 128 (4 waves)

  embed_kernel<<<gNE, TB, 0, stream>>>(x_idx, atom_emb, x, xh);

  for (int l = 0; l < NLAYER; ++l) {
    const float* Wl    = W + (size_t)l * EMB * EMB;
    const float* attSl = attS + (size_t)l * NH * CH;
    const float* attDl = attD + (size_t)l * NH * CH;
    const float* attEl = attE + (size_t)l * NH * CH;
    const float* biasl = bias + (size_t)l * EMB;
    const float* eembl = eembL + (size_t)l * 4 * EMB;
    const float* gml   = gamma + (size_t)l * EMB;
    const float* btl   = beta + (size_t)l * EMB;

    hipMemsetAsync(hist, 0, (size_t)NNODE * 4 * 4, stream);
    hipMemsetAsync(agg, 0, (size_t)NNODE * EMB * 4, stream);
    hipMemsetAsync(m_enc, 0, (size_t)NNODE * NH * 4, stream);   // 0 < ord(-inf)
    hipMemsetAsync(denom, 0, (size_t)NNODE * NH * 4, stream);
    hipMemsetAsync(bnsum, 0, EMB * 4, stream);
    hipMemsetAsync(bnsumsq, 0, EMB * 4, stream);

    packW_kernel<<<(4096 + TB - 1) / TB, TB, 0, stream>>>(Wl, Bp);
    gemm_xw_wmma<<<gW, 128, 0, stream>>>(xh, Bp, xl, NNODE);

    aet_kernel<<<1, 16, 0, stream>>>(eembl, attEl, aet);
    hist_kernel<<<gE, TB, 0, stream>>>(dst, eattr, hist);
    node_attn<<<gNH, TB, 0, stream>>>(xl, attSl, attDl, hist, aet, a_src, a_dst, a_loop);
    alpha_max<<<gEL, TB, 0, stream>>>(src, dst, eattr, a_src, a_dst, a_loop, aet,
                                      alpha_buf, m_enc);
    exp_sum<<<gEL, TB, 0, stream>>>(dst, alpha_buf, m_enc, denom);
    message<<<gMsg, TB, 0, stream>>>(src, dst, eattr, alpha_buf, denom, xl, eembl,
                                     hist, agg);
    bn_stats<<<dim3(EMB, (NNODE + TB - 1) / TB), TB, 0, stream>>>(agg, biasl, bnsum,
                                                                  bnsumsq);
    bn_apply<<<gNE, TB, 0, stream>>>(agg, biasl, bnsum, bnsumsq, gml, btl, x, xh);
  }

  predict<<<gE, TB, 0, stream>>>(x, src, dst, pW, pb, out);
}
